// CausalSelfAttention_65429531787879
// MI455X (gfx1250) — compile-verified, matrix operand order checked by reading, not run on hardware
//
#include <hip/hip_runtime.h>
#include <stddef.h>

typedef unsigned short u16;
typedef unsigned int   u32;

typedef __attribute__((ext_vector_type(16))) __bf16 v16bf;
typedef __attribute__((ext_vector_type(8)))  float  v8f;

#define BB   4
#define TT   2048
#define CC   2048
#define HH   16
#define KVH  4
#define HD   128

union FragU {
  v16bf v;
  struct { uint4 lo; uint4 hi; } q;
};

__device__ __forceinline__ u16 f2bf(float f) {
  u32 u = __float_as_uint(f);
  u32 r = u + 0x7fffu + ((u >> 16) & 1u);   // round-to-nearest-even
  return (u16)(r >> 16);
}
__device__ __forceinline__ float bf2f(u16 h) {
  return __uint_as_float(((u32)h) << 16);
}
__device__ __forceinline__ v8f wmma_bf16(const FragU& a, const FragU& b, v8f c) {
  return __builtin_amdgcn_wmma_f32_16x16x32_bf16(false, a.v, false, b.v,
                                                 (short)0, c, false, false);
}

// ---------------------------------------------------------------- converts
// 8 floats -> 8 bf16 per thread (b128 in / b128 store)
__global__ void cvt_f32_bf16_kernel(const float4* __restrict__ in,
                                    uint4* __restrict__ out, int n8) {
  int i = blockIdx.x * blockDim.x + threadIdx.x;
  if (i >= n8) return;
  float4 a = in[2 * i], b = in[2 * i + 1];
  union { u16 h[8]; uint4 u; } r;
  r.h[0] = f2bf(a.x); r.h[1] = f2bf(a.y); r.h[2] = f2bf(a.z); r.h[3] = f2bf(a.w);
  r.h[4] = f2bf(b.x); r.h[5] = f2bf(b.y); r.h[6] = f2bf(b.z); r.h[7] = f2bf(b.w);
  out[i] = r.u;
}

// W[K][N] fp32 -> Wt[N][K] bf16, 64x64 LDS-tiled transpose:
// coalesced b128 reads, LDS scatter, coalesced b128 writes.
__global__ __launch_bounds__(256) void wtrans64_kernel(const float* __restrict__ W,
                                                       u16* __restrict__ Wt,
                                                       int K, int N) {
  __shared__ u16 tile[64 * 72];           // stride 72 elems = 144B (16B-aligned)
  const int tid = threadIdx.x;
  const int k0 = blockIdx.y * 64, n0 = blockIdx.x * 64;
#pragma unroll
  for (int j = 0; j < 4; ++j) {           // 1024 float4 chunks
    int c  = tid + 256 * j;
    int r  = c >> 4;                      // k-row 0..63
    int c4 = (c & 15) * 4;                // n-col
    float4 v = *(const float4*)(W + (size_t)(k0 + r) * N + n0 + c4);
    tile[(c4 + 0) * 72 + r] = f2bf(v.x);
    tile[(c4 + 1) * 72 + r] = f2bf(v.y);
    tile[(c4 + 2) * 72 + r] = f2bf(v.z);
    tile[(c4 + 3) * 72 + r] = f2bf(v.w);
  }
  __syncthreads();
#pragma unroll
  for (int j = 0; j < 2; ++j) {           // 512 uint4 chunks
    int c  = tid + 256 * j;
    int rn = c >> 3;                      // n-row 0..63
    int c8 = (c & 7) * 8;                 // k-col
    *(uint4*)(Wt + (size_t)(n0 + rn) * K + k0 + c8) =
        *(const uint4*)&tile[rn * 72 + c8];
  }
}

// vlin[b][t][KVH*HD] -> Vt[b][kv][HD][TT], 64x64 LDS-tiled transpose
__global__ __launch_bounds__(256) void vtrans64_kernel(const u16* __restrict__ vlin,
                                                       u16* __restrict__ Vt) {
  __shared__ u16 tile[64 * 72];
  const int tid = threadIdx.x;
  const int t0 = blockIdx.x * 64;         // token tile
  const int d0 = blockIdx.y * 64;         // dim tile
  const int bz = blockIdx.z;              // b*KVH + kv
  const int b  = bz >> 2, kv = bz & 3;
  const u16* inb = vlin + (size_t)b * TT * KVH * HD + (size_t)kv * HD;
  u16* outb = Vt + (size_t)bz * HD * TT;
#pragma unroll
  for (int j = 0; j < 2; ++j) {           // 512 uint4 chunks in
    int c  = tid + 256 * j;
    int r  = c >> 3;                      // t-row
    int c8 = (c & 7) * 8;                 // d-col
    union { uint4 u; u16 h[8]; } w;
    w.u = *(const uint4*)(inb + (size_t)(t0 + r) * (KVH * HD) + d0 + c8);
#pragma unroll
    for (int e = 0; e < 8; ++e) tile[(c8 + e) * 72 + r] = w.h[e];
  }
  __syncthreads();
#pragma unroll
  for (int j = 0; j < 2; ++j) {           // 512 uint4 chunks out
    int c  = tid + 256 * j;
    int rd = c >> 3;                      // d-row
    int c8 = (c & 7) * 8;                 // t-col
    *(uint4*)(outb + (size_t)(d0 + rd) * TT + t0 + c8) =
        *(const uint4*)&tile[rd * 72 + c8];
  }
}

// ---------------------------------------------------------------- GEMM
// C[M][N] = A[M][2048] * Bt[N][2048]^T  (bf16 in, f32 accum)
// 256 thr = 8 waves; block tile 128x128; wave tile 64x32; K-step 32,
// double-buffered LDS, fetch-to-reg / commit-after-MAC software pipeline.
template <int N, bool F32OUT>
__global__ __launch_bounds__(256) void gemm_bf16_kernel(
    const u16* __restrict__ A, const u16* __restrict__ Bt,
    void* __restrict__ Cout) {
  constexpr int K = 2048;
  __shared__ u16 As[2][128 * 40];   // stride 40 elems = 80B: 16B-aligned, bank-spread
  __shared__ u16 Bs[2][128 * 40];
  const int tid  = threadIdx.x;
  const int wv   = tid >> 5, lane = tid & 31;
  const int m    = lane & 15, half = lane >> 4;
  const int wm   = wv >> 2,  wn   = wv & 3;     // 2 x 4 wave grid
  const int m0   = blockIdx.y * 128, n0 = blockIdx.x * 128;

  // staging geometry: 2 chunks/thread/matrix; (tid+256)&3 == tid&3
  const int srow = tid >> 2;              // 0..63
  const int scol = (tid & 3) * 8;
  uint4 ga0, ga1, gb0, gb1;

  auto fetch = [&](int k0) {
    ga0 = *(const uint4*)(A  + (size_t)(m0 + srow)      * K + k0 + scol);
    ga1 = *(const uint4*)(A  + (size_t)(m0 + srow + 64) * K + k0 + scol);
    gb0 = *(const uint4*)(Bt + (size_t)(n0 + srow)      * K + k0 + scol);
    gb1 = *(const uint4*)(Bt + (size_t)(n0 + srow + 64) * K + k0 + scol);
    if (k0 + 32 < K) {                    // warm L2 one step ahead
      __builtin_prefetch(A  + (size_t)(m0 + srow) * K + k0 + 32 + scol, 0, 0);
      __builtin_prefetch(Bt + (size_t)(n0 + srow) * K + k0 + 32 + scol, 0, 0);
    }
  };
  auto commit = [&](int buf) {
    *(uint4*)&As[buf][(size_t)srow        * 40 + scol] = ga0;
    *(uint4*)&As[buf][(size_t)(srow + 64) * 40 + scol] = ga1;
    *(uint4*)&Bs[buf][(size_t)srow        * 40 + scol] = gb0;
    *(uint4*)&Bs[buf][(size_t)(srow + 64) * 40 + scol] = gb1;
  };

  v8f acc[4][2];
#pragma unroll
  for (int mi = 0; mi < 4; ++mi)
#pragma unroll
    for (int ni = 0; ni < 2; ++ni)
#pragma unroll
      for (int e = 0; e < 8; ++e) acc[mi][ni][e] = 0.0f;

  fetch(0);
  commit(0);
  for (int k0 = 0; k0 < K; k0 += 32) {
    const int cur = (k0 >> 5) & 1;
    __syncthreads();                        // commit of cur visible; cur^1 free
    const bool more = (k0 + 32) < K;
    if (more) fetch(k0 + 32);               // HBM latency overlaps WMMAs below

    FragU af[4], bf2[2];
#pragma unroll
    for (int mi = 0; mi < 4; ++mi) {
      int r = wm * 64 + mi * 16 + m;
      af[mi].q.lo = *(const uint4*)&As[cur][r * 40 + half * 8];
      af[mi].q.hi = *(const uint4*)&As[cur][r * 40 + 16 + half * 8];
    }
#pragma unroll
    for (int ni = 0; ni < 2; ++ni) {
      int r = wn * 32 + ni * 16 + m;
      bf2[ni].q.lo = *(const uint4*)&Bs[cur][r * 40 + half * 8];
      bf2[ni].q.hi = *(const uint4*)&Bs[cur][r * 40 + 16 + half * 8];
    }
#pragma unroll
    for (int mi = 0; mi < 4; ++mi)
#pragma unroll
      for (int ni = 0; ni < 2; ++ni)
        acc[mi][ni] = wmma_bf16(af[mi], bf2[ni], acc[mi][ni]);

    if (more) commit(cur ^ 1);              // waits loadcnt only after the MACs
  }

#pragma unroll
  for (int mi = 0; mi < 4; ++mi)
#pragma unroll
    for (int ni = 0; ni < 2; ++ni)
#pragma unroll
      for (int r = 0; r < 8; ++r) {
        int row = m0 + wm * 64 + mi * 16 + r + 8 * half;  // C: M=r+8*(lane/16)
        int col = n0 + wn * 32 + ni * 16 + m;             //    N=lane%16
        float v = acc[mi][ni][r];
        if constexpr (F32OUT) ((float*)Cout)[(size_t)row * N + col] = v;
        else                  ((u16*)Cout)[(size_t)row * N + col] = f2bf(v);
      }
}

// ---------------------------------------------------------------- RoPE + reshape
__global__ void rope_reshape_kernel(const u16* __restrict__ lin,
                                    u16* __restrict__ out,
                                    const int* __restrict__ posp,
                                    int nh, int total) {
  int i = blockIdx.x * blockDim.x + threadIdx.x;
  if (i >= total) return;
  int d   = i & (HD - 1);
  int rem = i >> 7;
  int h   = rem % nh; rem /= nh;
  int t   = rem % TT;
  int b   = rem / TT;
  int p   = posp[0] + t;
  size_t in_base = ((size_t)(b * TT + t) * nh + h) * HD;
  float xv = bf2f(lin[in_base + d]);
  float xp = (d < HD / 2) ? -bf2f(lin[in_base + d + HD / 2])
                          :  bf2f(lin[in_base + d - HD / 2]);
  // inv_freq[d//2] = 10000^(-(d&~1)/HD)
  float freq = __powf(10000.0f, -(float)(d & ~1) * (1.0f / HD));
  float ang  = (float)p * freq;
  float s, c;
  __sincosf(ang, &s, &c);
  out[((size_t)(b * nh + h) * TT + t) * HD + d] = f2bf(xv * c + xp * s);
}

// ---------------------------------------------------------------- attention
// Q[b][h][t][HD], Kc[b][kv][t][HD], Vt[b][kv][HD][t] -> AO[b*TT+t][CC]
// One block = (b, h, 128-query tile); 8 waves * 16 query rows each;
// 32-key tiles (one WMMA K), double-buffered Ks/Vs, flash online softmax.
__global__ __launch_bounds__(256) void attn_kernel(
    const u16* __restrict__ Q, const u16* __restrict__ Kc,
    const u16* __restrict__ Vt, u16* __restrict__ AO) {
  __shared__ u16 Ks[2][32 * 136];   // 32 keys x 128 d, stride 136 (272B)
  __shared__ u16 Vs[2][128 * 40];   // 128 d x 32 keys, stride 40
  __shared__ u16 Ps[8 * 16 * 40];   // per-wave P relayout scratch
  const int tid  = threadIdx.x;
  const int wv   = tid >> 5, lane = tid & 31;
  const int m    = lane & 15, half = lane >> 4;
  const int b    = blockIdx.z, h = blockIdx.y;
  const int kvh  = h >> 2;                 // H/KV = 4
  const int q0   = blockIdx.x * 128;
  const int qr   = q0 + wv * 16;

  FragU aq[4];
  const u16* qbase = Q + ((size_t)(b * HH + h) * TT + (qr + m)) * HD;
#pragma unroll
  for (int kb = 0; kb < 4; ++kb) {
    aq[kb].q.lo = *(const uint4*)(qbase + kb * 32 + half * 8);
    aq[kb].q.hi = *(const uint4*)(qbase + kb * 32 + 16 + half * 8);
  }

  const u16* kbase = Kc + (size_t)(b * KVH + kvh) * TT * HD;
  const u16* vbase = Vt + (size_t)(b * KVH + kvh) * HD * TT;

  // staging geometry
  const int krow = tid >> 4;            // 0..15   (Ks rows, +16 second chunk)
  const int kcol = (tid & 15) * 8;
  const int vrow = tid >> 2;            // 0..63   (Vs rows, +64 second chunk)
  const int vcol = (tid & 3) * 8;
  uint4 gk0, gk1, gv0, gv1;
  const int nkt = q0 / 32 + 4;          // causal tile count for this block

  auto fetch = [&](int kt) {
    gk0 = *(const uint4*)(kbase + (size_t)(kt * 32 + krow)      * HD + kcol);
    gk1 = *(const uint4*)(kbase + (size_t)(kt * 32 + krow + 16) * HD + kcol);
    gv0 = *(const uint4*)(vbase + (size_t)vrow        * TT + kt * 32 + vcol);
    gv1 = *(const uint4*)(vbase + (size_t)(vrow + 64) * TT + kt * 32 + vcol);
    if (kt + 1 < nkt) {                 // warm L2 one key tile ahead
      __builtin_prefetch(kbase + (size_t)(kt * 32 + 32 + krow) * HD + kcol, 0, 0);
      __builtin_prefetch(vbase + (size_t)vrow * TT + kt * 32 + 32 + vcol, 0, 0);
    }
  };
  auto commit = [&](int buf) {
    *(uint4*)&Ks[buf][(size_t)krow        * 136 + kcol] = gk0;
    *(uint4*)&Ks[buf][(size_t)(krow + 16) * 136 + kcol] = gk1;
    *(uint4*)&Vs[buf][(size_t)vrow        * 40 + vcol]  = gv0;
    *(uint4*)&Vs[buf][(size_t)(vrow + 64) * 40 + vcol]  = gv1;
  };

  v8f o[8];
  float rm[8], rl[8];
#pragma unroll
  for (int dt = 0; dt < 8; ++dt)
#pragma unroll
    for (int e = 0; e < 8; ++e) o[dt][e] = 0.0f;
#pragma unroll
  for (int r = 0; r < 8; ++r) { rm[r] = -3.0e38f; rl[r] = 1.0e-30f; }

  fetch(0);
  commit(0);
  for (int kt = 0; kt < nkt; ++kt) {
    const int cur = kt & 1;
    __syncthreads();                     // staged cur visible; cur^1 free
    const bool more = (kt + 1) < nkt;
    if (more) fetch(kt + 1);             // overlaps the tile's WMMAs

    if (kt * 32 <= qr + 15) {            // wave-uniform causal skip
      v8f s0, s1;
#pragma unroll
      for (int e = 0; e < 8; ++e) { s0[e] = 0.0f; s1[e] = 0.0f; }
#pragma unroll
      for (int kb = 0; kb < 4; ++kb) {   // S = Q * K^T over D=128
        FragU bk0, bk1;
        bk0.q.lo = *(const uint4*)&Ks[cur][m * 136 + kb * 32 + half * 8];
        bk0.q.hi = *(const uint4*)&Ks[cur][m * 136 + kb * 32 + 16 + half * 8];
        bk1.q.lo = *(const uint4*)&Ks[cur][(16 + m) * 136 + kb * 32 + half * 8];
        bk1.q.hi = *(const uint4*)&Ks[cur][(16 + m) * 136 + kb * 32 + 16 + half * 8];
        s0 = wmma_bf16(aq[kb], bk0, s0);
        s1 = wmma_bf16(aq[kb], bk1, s1);
      }

      const float sc = 0.088388347648318447f;  // 1/sqrt(128)
#pragma unroll
      for (int r = 0; r < 8; ++r) {
        int qi = qr + r + 8 * half;
        float v0 = s0[r] * sc; if (kt * 32 + m > qi)      v0 = -1.0e30f;
        float v1 = s1[r] * sc; if (kt * 32 + 16 + m > qi) v1 = -1.0e30f;
        float mr = fmaxf(v0, v1);        // row max over the 16-lane half
        mr = fmaxf(mr, __shfl_xor(mr, 1, 32));
        mr = fmaxf(mr, __shfl_xor(mr, 2, 32));
        mr = fmaxf(mr, __shfl_xor(mr, 4, 32));
        mr = fmaxf(mr, __shfl_xor(mr, 8, 32));
        float mnew  = fmaxf(rm[r], mr);
        float scale = __expf(rm[r] - mnew);
        float e0 = __expf(v0 - mnew);
        float e1 = __expf(v1 - mnew);
        float rs = e0 + e1;
        rs += __shfl_xor(rs, 1, 32);
        rs += __shfl_xor(rs, 2, 32);
        rs += __shfl_xor(rs, 4, 32);
        rs += __shfl_xor(rs, 8, 32);
        rl[r] = rl[r] * scale + rs;
        rm[r] = mnew;
#pragma unroll
        for (int dt = 0; dt < 8; ++dt) o[dt][r] *= scale;
        // P relayout (C-layout -> A-layout), wave-private LDS scratch
        Ps[wv * 640 + (r + 8 * half) * 40 + m]      = f2bf(e0);
        Ps[wv * 640 + (r + 8 * half) * 40 + 16 + m] = f2bf(e1);
      }
      // LDS executes per-wave in order (ISA ch.11); fence compiler reordering
      __builtin_amdgcn_wave_barrier();

      FragU ap;
      ap.q.lo = *(const uint4*)&Ps[wv * 640 + m * 40 + half * 8];
      ap.q.hi = *(const uint4*)&Ps[wv * 640 + m * 40 + 16 + half * 8];
#pragma unroll
      for (int dt = 0; dt < 8; ++dt) {   // O += P * V  (K=32)
        FragU bvf;
        int n = dt * 16 + m;
        bvf.q.lo = *(const uint4*)&Vs[cur][n * 40 + half * 8];
        bvf.q.hi = *(const uint4*)&Vs[cur][n * 40 + 16 + half * 8];
        o[dt] = wmma_bf16(ap, bvf, o[dt]);
      }
    }

    if (more) commit(cur ^ 1);           // loadcnt wait lands after the MACs
  }

  float inv[8];
#pragma unroll
  for (int r = 0; r < 8; ++r) inv[r] = 1.0f / rl[r];
#pragma unroll
  for (int dt = 0; dt < 8; ++dt)
#pragma unroll
    for (int r = 0; r < 8; ++r) {
      AO[((size_t)(b * TT) + qr + r + 8 * half) * CC + h * HD + dt * 16 + m] =
          f2bf(o[dt][r] * inv[r]);
    }
}

// ---------------------------------------------------------------- host
extern "C" void kernel_launch(void* const* d_in, const int* in_sizes, int n_in,
                              void* d_out, int out_size, void* d_ws, size_t ws_size,
                              hipStream_t stream) {
  (void)in_sizes; (void)n_in; (void)out_size; (void)ws_size;
  const float* x  = (const float*)d_in[0];
  const float* Wq = (const float*)d_in[1];
  const float* Wk = (const float*)d_in[2];
  const float* Wv = (const float*)d_in[3];
  const float* Wo = (const float*)d_in[4];
  const int*   pos = (const int*)d_in[5];
  float* out = (float*)d_out;

  char* w = (char*)d_ws;
  size_t off = 0;
  auto alloc = [&](size_t elems) {
    u16* p = (u16*)(w + off);
    off += ((elems * 2 + 255) & ~(size_t)255);
    return p;
  };
  u16* xb   = alloc(16777216);   // x bf16 [8192][2048]; reused as AO later
  u16* WqT  = alloc(4194304);    // [2048][2048]
  u16* WkT  = alloc(1048576);    // [512][2048]
  u16* WvT  = alloc(1048576);
  u16* WoT  = alloc(4194304);
  u16* qlin = alloc(16777216);
  u16* klin = alloc(4194304);
  u16* vlin = alloc(4194304);
  u16* Qh   = alloc(16777216);   // [B][H][T][D]
  u16* Kh   = alloc(4194304);    // [B][KV][T][D]
  u16* Vt   = alloc(4194304);    // [B][KV][D][T]
  u16* AO   = xb;

  cvt_f32_bf16_kernel<<<8192, 256, 0, stream>>>((const float4*)x, (uint4*)xb, 2097152);
  wtrans64_kernel<<<dim3(32, 32), 256, 0, stream>>>(Wq, WqT, 2048, 2048);
  wtrans64_kernel<<<dim3( 8, 32), 256, 0, stream>>>(Wk, WkT, 2048, 512);
  wtrans64_kernel<<<dim3( 8, 32), 256, 0, stream>>>(Wv, WvT, 2048, 512);
  wtrans64_kernel<<<dim3(32, 32), 256, 0, stream>>>(Wo, WoT, 2048, 2048);

  gemm_bf16_kernel<2048, false><<<dim3(16, 64), 256, 0, stream>>>(xb, WqT, qlin);
  gemm_bf16_kernel< 512, false><<<dim3( 4, 64), 256, 0, stream>>>(xb, WkT, klin);
  gemm_bf16_kernel< 512, false><<<dim3( 4, 64), 256, 0, stream>>>(xb, WvT, vlin);

  rope_reshape_kernel<<<65536, 256, 0, stream>>>(qlin, Qh, pos, 16, 16777216);
  rope_reshape_kernel<<<16384, 256, 0, stream>>>(klin, Kh, pos,  4,  4194304);
  vtrans64_kernel<<<dim3(32, 2, 16), 256, 0, stream>>>(vlin, Vt);

  attn_kernel<<<dim3(16, 16, 4), 256, 0, stream>>>(Qh, Kh, Vt, AO);

  gemm_bf16_kernel<2048, true><<<dim3(16, 64), 256, 0, stream>>>(AO, WoT, out);
}
